// TransformerEncoder_44796508897506
// MI455X (gfx1250) — matrix operand order repaired and checked by reading, not compiled, not run on hardware
//
#include <hip/hip_runtime.h>
#include <hip/hip_bf16.h>
#include <stdint.h>

// Problem constants (B,S,E,H,L) = (4,2048,1024,16,4), D = 64
#define B_   4
#define S_   2048
#define E_   1024
#define H_   16
#define D_   64
#define L_   4
#define BS_  (B_*S_)
#define EPS_ 1e-5f

typedef __attribute__((ext_vector_type(16))) __bf16        v16bf;
typedef __attribute__((ext_vector_type(8)))  float         v8f;
typedef __attribute__((ext_vector_type(8)))  unsigned int  v8u;

__device__ __forceinline__ unsigned short f32_to_bf16(float f) {
  unsigned int u = __float_as_uint(f);
  u += 0x7fffu + ((u >> 16) & 1u);           // round to nearest even
  return (unsigned short)(u >> 16);
}
__device__ __forceinline__ unsigned int pack2_bf16(float a, float b) {
  return (unsigned int)f32_to_bf16(a) | ((unsigned int)f32_to_bf16(b) << 16);
}

__device__ __forceinline__ v8f wmma_bf16(v16bf a, v16bf b, v8f c) {
  // D(f32 16x16) = A(bf16 16x32) * B(bf16 32x16) + C
  return __builtin_amdgcn_wmma_f32_16x16x32_bf16(
      /*neg_a=*/false, a, /*neg_b=*/false, b,
      /*c_mod=*/(short)0, c, /*reuse_a=*/false, /*reuse_b=*/false);
}

// CDNA5 async memory->LDS copy, 16B per lane, tracked by ASYNCcnt (ISA 08 §4).
// lds_addr = low 32 bits of generic pointer to __shared__ (LDS aperture keeps
// the byte offset in addr[31:0]); gaddr in a VGPR pair (GV addressing mode).
__device__ __forceinline__ void async_copy_b128(unsigned lds_addr, const void* gptr) {
  asm volatile("global_load_async_to_lds_b128 %0, %1, off"
               :: "v"(lds_addr), "v"((unsigned long long)(uintptr_t)gptr)
               : "memory");
}
__device__ __forceinline__ void wait_asynccnt0() {
  asm volatile("s_wait_asynccnt 0" ::: "memory");
}

// A-operand 16x32 bf16 fragment per CDNA5 ISA layout:
//  lane m = L&15, half = L>>4; VGPR j(0..3): K = half*8 + 2j..2j+1, VGPR 4..7: +16
//  -> per lane: two contiguous 16B reads (K = half*8..+7 and 16+half*8..+7).
__device__ __forceinline__ v16bf load_a_frag(const unsigned short* base, int stride,
                                             int lane, int koff) {
  int m = lane & 15, half = lane >> 4;
  const unsigned short* p = base + (size_t)m * stride + koff + half * 8;
  uint4 x = *(const uint4*)p;
  uint4 y = *(const uint4*)(p + 16);
  v8u r; r[0]=x.x; r[1]=x.y; r[2]=x.z; r[3]=x.w; r[4]=y.x; r[5]=y.y; r[6]=y.z; r[7]=y.w;
  return __builtin_bit_cast(v16bf, r);
}
// B-operand 32x16 bf16 fragment (K x N), base points at Bt row-major [N][K]:
//  lane n = L&15, half = L>>4 holds K = half*8..+7 and 16+half*8..+7 (same as A).
__device__ __forceinline__ v16bf load_b_frag(const unsigned short* base, int stride,
                                             int lane, int koff) {
  int n = lane & 15, half = lane >> 4;
  const unsigned short* p = base + (size_t)n * stride + koff + half * 8;
  uint4 x = *(const uint4*)p;
  uint4 y = *(const uint4*)(p + 16);
  v8u r; r[0]=x.x; r[1]=x.y; r[2]=x.z; r[3]=x.w; r[4]=y.x; r[5]=y.y; r[6]=y.z; r[7]=y.w;
  return __builtin_bit_cast(v16bf, r);
}

// ---------------------------------------------------------------------------
// C[M,E] = A[M,E](bf16) @ W[E,E]^T (f32 -> bf16 on stage) + bias
// 256 threads = 8 waves (4x2), block tile 128x128, wave tile 32x64, K chunk 64.
// A tile staged with GLOBAL_LOAD_ASYNC_TO_LDS_B128; W converted f32->bf16 inline.
// ---------------------------------------------------------------------------
template <bool OUT_BF16>
__global__ __launch_bounds__(256) void gemm_xwt(
    const unsigned short* __restrict__ Ag, const float* __restrict__ Wg,
    const float* __restrict__ bias, unsigned short* __restrict__ Ob,
    float* __restrict__ Of) {
  __shared__ __align__(16) unsigned short As[128][72];  // +8 bf16 pad: conflict-free
  __shared__ __align__(16) unsigned short Ws[128][72];

  const int tid  = threadIdx.x;
  const int lane = tid & 31, wave = tid >> 5;
  const int wm = wave >> 1, wn = wave & 1;   // 4 x 2 waves -> 32x64 per wave
  const int n0 = blockIdx.x * 128;
  const int m0 = blockIdx.y * 128;

  v8f acc[2][4];
  for (int i = 0; i < 2; i++)
    for (int j = 0; j < 4; j++)
      for (int e = 0; e < 8; e++) acc[i][j][e] = 0.f;

  for (int k0 = 0; k0 < E_; k0 += 64) {
    // Stage A tile 128x64 bf16: async memory->LDS, 16B per lane-issue
    for (int i = 0; i < 4; i++) {
      int idx = tid + i * 256;            // 0..1023 : 16B chunks
      int r = idx >> 3, c8 = idx & 7;     // 8 chunks of 8 bf16 per row
      async_copy_b128((unsigned)(uintptr_t)&As[r][c8 * 8],
                      Ag + (size_t)(m0 + r) * E_ + k0 + c8 * 8);
    }
    // Stage W tile 128x64, convert f32 -> bf16 in flight
    for (int i = 0; i < 8; i++) {
      int idx = tid + i * 256;
      int r = idx >> 4, c4 = idx & 15;
      float4 w = *(const float4*)(Wg + (size_t)(n0 + r) * E_ + k0 + c4 * 4);
      uint2 pk; pk.x = pack2_bf16(w.x, w.y); pk.y = pack2_bf16(w.z, w.w);
      *(uint2*)&Ws[r][c4 * 4] = pk;
    }
    if (k0 + 64 < E_) {  // warm caches for next W tile (global_prefetch_b8)
      __builtin_prefetch(Wg + (size_t)(n0 + (tid >> 1)) * E_ + k0 + 64 + (tid & 1) * 32, 0, 3);
    }
    wait_asynccnt0();   // async A-tile landed in LDS
    __syncthreads();
    for (int kk = 0; kk < 64; kk += 32) {
      v16bf af[2], bfg[4];
      af[0] = load_a_frag(&As[wm * 32][0],      72, lane, kk);
      af[1] = load_a_frag(&As[wm * 32 + 16][0], 72, lane, kk);
      for (int j = 0; j < 4; j++)
        bfg[j] = load_b_frag(&Ws[wn * 64 + j * 16][0], 72, lane, kk);
      for (int i = 0; i < 2; i++)
        for (int j = 0; j < 4; j++) acc[i][j] = wmma_bf16(af[i], bfg[j], acc[i][j]);
    }
    __syncthreads();
  }

  // Epilogue: C layout -> lane col n=L&15, VGPR g -> row (L>>4)*8 + g
  const int half = lane >> 4, nn = lane & 15;
  for (int i = 0; i < 2; i++)
    for (int j = 0; j < 4; j++) {
      int col = n0 + wn * 64 + j * 16 + nn;
      float bb = bias[col];
      for (int g = 0; g < 8; g++) {
        int row = m0 + wm * 32 + i * 16 + half * 8 + g;
        float v = acc[i][j][g] + bb;
        if (OUT_BF16) Ob[(size_t)row * E_ + col] = f32_to_bf16(v);
        else          Of[(size_t)row * E_ + col] = v;
      }
    }
}

// ---------------------------------------------------------------------------
// Flash attention: one block (128 thr = 4 waves) per (b,h, 32 query rows).
// Streams 32-key chunks: S = QK^T (WMMA), online softmax in LDS, O += P@V (WMMA).
// q/k/v/o are bf16 [B,S,H,D] views of [B,S,E].
// ---------------------------------------------------------------------------
__global__ __launch_bounds__(128) void attn_kernel(
    const unsigned short* __restrict__ Qg, const unsigned short* __restrict__ Kg,
    const unsigned short* __restrict__ Vg, const int* __restrict__ maskg,
    unsigned short* __restrict__ Og) {
  __shared__ __align__(16) float          sc[32][36];   // score tile f32
  __shared__ __align__(16) unsigned short Pt[32][40];   // exp(scores) bf16
  __shared__ __align__(16) unsigned short Vt[64][40];   // V^T chunk [d][key]
  __shared__ float rowm[32], rowl[32], rowalpha[32];
  __shared__ float pmax[32][4], psum[32][4];

  const int tid  = threadIdx.x;
  const int lane = tid & 31, wave = tid >> 5;
  const int wm = wave >> 1, wn = wave & 1;  // 2x2 waves over 32x32 tile
  const int q0 = blockIdx.x * 32;
  const int b  = blockIdx.y / H_, h = blockIdx.y % H_;
  const int half = lane >> 4, nn = lane & 15;
  const int srow = tid >> 2, q4 = tid & 3;  // 4 threads per score row
  const float inv_scale = 0.03125f;         // 1/sqrt(E)

  if (tid < 32) { rowm[tid] = -3.0e38f; rowl[tid] = 0.f; }
  __syncthreads();

  // Q fragments straight from global: contraction dim d is contiguous.
  const unsigned short* qbase = Qg + (size_t)(b * S_ + q0 + wm * 16) * E_ + h * D_;
  v16bf qf0 = load_a_frag(qbase, E_, lane, 0);
  v16bf qf1 = load_a_frag(qbase, E_, lane, 32);

  v8f oacc[2];
  for (int j = 0; j < 2; j++)
    for (int e = 0; e < 8; e++) oacc[j][e] = 0.f;

  for (int key0 = 0; key0 < S_; key0 += 32) {
    // --- scores: S[32x32] = Q[32x64] @ K^T ---
    const unsigned short* kbase =
        Kg + (size_t)(b * S_ + key0 + wn * 16) * E_ + h * D_;
    v16bf kf0 = load_b_frag(kbase, E_, lane, 0);
    v16bf kf1 = load_b_frag(kbase, E_, lane, 32);
    v8f s; for (int e = 0; e < 8; e++) s[e] = 0.f;
    s = wmma_bf16(qf0, kf0, s);
    s = wmma_bf16(qf1, kf1, s);
    for (int g = 0; g < 8; g++) sc[wm * 16 + half * 8 + g][wn * 16 + nn] = s[g];
    __syncthreads();

    // --- phase A: mask + scale, per-thread max over 8 cols ---
    float val[8]; float lmax = -3.0e38f;
    for (int c = 0; c < 8; c++) {
      int col = q4 * 8 + c;
      float e = sc[srow][col];
      int mk = maskg[b * S_ + key0 + col];
      float x = ((mk == 1) ? -1e-10f : e) * inv_scale;  // reference masks pre-scale
      val[c] = x;
      lmax = fmaxf(lmax, x);
    }
    pmax[srow][q4] = lmax;
    __syncthreads();

    // --- phase B: online max/exp, write P(bf16), stage V^T ---
    float m_old = rowm[srow];
    float m_new = fmaxf(
        fmaxf(fmaxf(pmax[srow][0], pmax[srow][1]), fmaxf(pmax[srow][2], pmax[srow][3])),
        m_old);
    float lsum = 0.f;
    for (int c = 0; c < 8; c++) {
      float p = __expf(val[c] - m_new);
      lsum += p;
      Pt[srow][q4 * 8 + c] = f32_to_bf16(p);
    }
    psum[srow][q4] = lsum;
    if (q4 == 0) { rowalpha[srow] = __expf(m_old - m_new); rowm[srow] = m_new; }
    for (int j = 0; j < 4; j++) {  // V chunk 32x64 -> transposed Vt[d][key]
      int e = tid + j * 128;
      int key = e >> 4, d4 = e & 15;
      uint2 vv = *(const uint2*)(Vg + (size_t)(b * S_ + key0 + key) * E_ + h * D_ + d4 * 4);
      Vt[d4 * 4 + 0][key] = (unsigned short)(vv.x & 0xffffu);
      Vt[d4 * 4 + 1][key] = (unsigned short)(vv.x >> 16);
      Vt[d4 * 4 + 2][key] = (unsigned short)(vv.y & 0xffffu);
      Vt[d4 * 4 + 3][key] = (unsigned short)(vv.y >> 16);
    }
    __syncthreads();

    // --- phase C: rescale O, O += P @ V ---
    if (q4 == 0) {
      rowl[srow] = rowl[srow] * rowalpha[srow] +
                   psum[srow][0] + psum[srow][1] + psum[srow][2] + psum[srow][3];
    }
    for (int j = 0; j < 2; j++)
      for (int g = 0; g < 8; g++) oacc[j][g] *= rowalpha[wm * 16 + half * 8 + g];
    v16bf pf = load_a_frag(&Pt[wm * 16][0], 40, lane, 0);
    for (int j = 0; j < 2; j++) {
      v16bf vf = load_b_frag(&Vt[wn * 32 + j * 16][0], 40, lane, 0);
      oacc[j] = wmma_bf16(pf, vf, oacc[j]);
    }
    __syncthreads();
  }

  // finalize: divide by l, store bf16 o[b,s,h,d]
  for (int j = 0; j < 2; j++)
    for (int g = 0; g < 8; g++) {
      int rrel = wm * 16 + half * 8 + g;
      float v = oacc[j][g] / rowl[rrel];
      Og[(size_t)(b * S_ + q0 + rrel) * E_ + h * D_ + wn * 32 + j * 16 + nn] =
          f32_to_bf16(v);
    }
}

// ---------------------------------------------------------------------------
// y = LayerNorm(oproj + x) * gamma + beta ; emits fp32 (residual stream) + bf16
// ---------------------------------------------------------------------------
__global__ __launch_bounds__(256) void ln_kernel(
    const float* __restrict__ oproj, const float* __restrict__ xin,
    const float* __restrict__ gamma, const float* __restrict__ beta,
    float* __restrict__ yout, unsigned short* __restrict__ ybf) {
  __shared__ float r1[256], r2[256];
  const int row = blockIdx.x, t = threadIdx.x;
  float v[4]; float s = 0.f, s2 = 0.f;
  for (int i = 0; i < 4; i++) {
    int idx = t + i * 256;
    float y = oproj[(size_t)row * E_ + idx] + xin[(size_t)row * E_ + idx];
    v[i] = y; s += y; s2 += y * y;
  }
  r1[t] = s; r2[t] = s2; __syncthreads();
  for (int off = 128; off > 0; off >>= 1) {
    if (t < off) { r1[t] += r1[t + off]; r2[t] += r2[t + off]; }
    __syncthreads();
  }
  float mu  = r1[0] * (1.f / E_);
  float var = r2[0] * (1.f / E_) - mu * mu;
  float rs  = rsqrtf(var + EPS_);
  for (int i = 0; i < 4; i++) {
    int idx = t + i * 256;
    float o = (v[i] - mu) * rs * gamma[idx] + beta[idx];
    yout[(size_t)row * E_ + idx] = o;
    ybf[(size_t)row * E_ + idx]  = f32_to_bf16(o);
  }
}

__global__ __launch_bounds__(256) void cvt_bf16_kernel(
    const float* __restrict__ in, unsigned short* __restrict__ out, int n) {
  for (int i = blockIdx.x * blockDim.x + threadIdx.x; i < n;
       i += gridDim.x * blockDim.x)
    out[i] = f32_to_bf16(in[i]);
}

// ---------------------------------------------------------------------------
extern "C" void kernel_launch(void* const* d_in, const int* in_sizes, int n_in,
                              void* d_out, int out_size, void* d_ws, size_t ws_size,
                              hipStream_t stream) {
  (void)in_sizes; (void)n_in; (void)out_size; (void)ws_size;
  const float* value = (const float*)d_in[0];
  const int*   mask  = (const int*)d_in[1];
  const float* Wq = (const float*)d_in[2];
  const float* bq = (const float*)d_in[3];
  const float* Wk = (const float*)d_in[4];
  const float* bk = (const float*)d_in[5];
  const float* Wv = (const float*)d_in[6];
  const float* bv = (const float*)d_in[7];
  const float* Wo = (const float*)d_in[8];
  const float* bo = (const float*)d_in[9];
  const float* gamma = (const float*)d_in[10];
  const float* beta  = (const float*)d_in[11];

  char* ws = (char*)d_ws;
  const size_t sBF = (size_t)BS_ * E_ * sizeof(unsigned short);  // 16 MB
  unsigned short* xbf = (unsigned short*)(ws);
  unsigned short* qbf = (unsigned short*)(ws + sBF);
  unsigned short* kbf = (unsigned short*)(ws + 2 * sBF);
  unsigned short* vbf = (unsigned short*)(ws + 3 * sBF);
  unsigned short* obf = (unsigned short*)(ws + 4 * sBF);
  float* oproj = (float*)(ws + 5 * sBF);
  float* xf    = (float*)(ws + 5 * sBF + (size_t)BS_ * E_ * sizeof(float));
  // total workspace: 5*16MB + 2*32MB = 144 MB

  cvt_bf16_kernel<<<512, 256, 0, stream>>>(value, xbf, BS_ * E_);

  dim3 gg(E_ / 128, BS_ / 128);    // 8 x 64 blocks
  dim3 ga(S_ / 32, B_ * H_);       // 64 x 64 blocks
  const float* xin = value;
  for (int l = 0; l < L_; l++) {
    size_t wo = (size_t)l * E_ * E_, bf = (size_t)l * E_;
    gemm_xwt<true ><<<gg, 256, 0, stream>>>(xbf, Wq + wo, bq + bf, qbf, nullptr);
    gemm_xwt<true ><<<gg, 256, 0, stream>>>(xbf, Wk + wo, bk + bf, kbf, nullptr);
    gemm_xwt<true ><<<gg, 256, 0, stream>>>(xbf, Wv + wo, bv + bf, vbf, nullptr);
    attn_kernel<<<ga, 128, 0, stream>>>(qbf, kbf, vbf, mask, obf);
    gemm_xwt<false><<<gg, 256, 0, stream>>>(obf, Wo + wo, bo + bf, nullptr, oproj);
    float* ydst = (l == L_ - 1) ? (float*)d_out : xf;
    ln_kernel<<<BS_, 256, 0, stream>>>(oproj, xin, gamma + bf, beta + bf, ydst, xbf);
    xin = ydst;
  }
}
